// Seq2Seq_53051436040587
// MI455X (gfx1250) — matrix operand order repaired and checked by reading, not compiled, not run on hardware
//
#include <hip/hip_runtime.h>
#include <math.h>

// ---------------- problem constants ----------------
constexpr int VOCAB = 32000;
constexpr int E     = 256;
constexpr int HE    = 256;   // H_ENC
constexpr int HD    = 512;   // H_DEC
constexpr int B_    = 32;
constexpr int S_    = 400;
constexpr int T_    = 100;

// ---------------- WMMA types / helpers ----------------
typedef __attribute__((ext_vector_type(16))) __bf16          v16bf;
typedef __attribute__((ext_vector_type(8)))  float           v8f;
typedef __attribute__((ext_vector_type(8)))  unsigned short  ushort8;
typedef __attribute__((ext_vector_type(16))) unsigned short  ushort16;

__device__ inline unsigned short f2bf(float f) {
    unsigned u = __builtin_bit_cast(unsigned, f);
    unsigned r = u + 0x7FFFu + ((u >> 16) & 1u);   // round-to-nearest-even
    return (unsigned short)(r >> 16);
}

__device__ inline v16bf mk16(ushort8 lo, ushort8 hi) {
    ushort16 u;
#pragma unroll
    for (int i = 0; i < 8; ++i) { u[i] = lo[i]; u[i + 8] = hi[i]; }
    return __builtin_bit_cast(v16bf, u);
}

// A fragment: 16x32 bf16 tile of row-major X (ld elements per row).
// lane 0-15: M=lane, K = kb+{0..7, 16..23}; lane 16-31: M=lane-16, K = kb+{8..15, 24..31}
__device__ inline v16bf load_a(const unsigned short* X, int ld, int m0, int kb, int lane) {
    int row = m0 + (lane & 15);
    int k2  = kb + ((lane >> 4) << 3);
    const ushort8* p = (const ushort8*)(X + (long)row * ld + k2);
    return mk16(p[0], p[2]);          // chunks at K=k2 and K=k2+16
}

// B fragment: 32x16 bf16 tile where B[k,n] = W[n,k] (W row-major (N,K), ld=K).
// lane 0-15: N=lane, K = kb+{0..7, 16..23}; lane 16-31: N=lane-16, K = kb+{8..15, 24..31}
__device__ inline v16bf load_b(const unsigned short* W, int ld, int n0, int kb, int lane) {
    int col = n0 + (lane & 15);
    int k2  = kb + ((lane >> 4) << 3);
    const ushort8* p = (const ushort8*)(W + (long)col * ld + k2);
    return mk16(p[0], p[2]);
}

__device__ inline v8f wmma_bf16(v8f acc, v16bf a, v16bf b) {
    return __builtin_amdgcn_wmma_f32_16x16x32_bf16(
        /*neg_a=*/false, a, /*neg_b=*/false, b,
        /*c_mod=*/(short)0, acc, /*reuse_a=*/false, /*reuse_b=*/false);
}

__device__ inline float sigmoidf_(float x) { return 1.f / (1.f + __expf(-x)); }

// ---------------- utility kernels ----------------
__global__ void k_f32_to_bf16(const float* __restrict__ src,
                              unsigned short* __restrict__ dst, long n) {
    for (long i = (long)blockIdx.x * blockDim.x + threadIdx.x; i < n;
         i += (long)gridDim.x * blockDim.x)
        dst[i] = f2bf(src[i]);
}

// ids: (B, L) int32; out: (L, B, E) bf16
__global__ void k_embed(const int* __restrict__ ids, const float* __restrict__ tab,
                        unsigned short* __restrict__ out, int L, long total) {
    for (long i = (long)blockIdx.x * blockDim.x + threadIdx.x; i < total;
         i += (long)gridDim.x * blockDim.x) {
        int  e = (int)(i & 255);
        long r = i >> 8;
        int  b = (int)(r % B_);
        int  l = (int)(r / B_);
        int tok = ids[(long)b * L + l];
        out[i] = f2bf(tab[(long)tok * E + e]);
    }
}

// ---------------- encoder: persistent, blockIdx.x = direction ----------------
// LDS: h_f32(32KB) | h_bf16(16KB) | g_rz(64KB) | g_in(32KB) | g_hn(32KB) = 176KB
__global__ __launch_bounds__(512) void k_encoder(
    const unsigned short* __restrict__ xs,                 // (S,B,E) bf16
    const unsigned short* __restrict__ wih0, const unsigned short* __restrict__ whh0,
    const float* __restrict__ bih0, const float* __restrict__ bhh0,
    const unsigned short* __restrict__ wih1, const unsigned short* __restrict__ whh1,
    const float* __restrict__ bih1, const float* __restrict__ bhh1,
    float* __restrict__ enhy, unsigned short* __restrict__ enhy_bf) // (B,S,2*HE)
{
    extern __shared__ char smem[];
    float*          h_f  = (float*)(smem);
    unsigned short* h_b  = (unsigned short*)(smem + 32768);
    float*          g_rz = (float*)(smem + 49152);
    float*          g_in = (float*)(smem + 114688);
    float*          g_hn = (float*)(smem + 147456);

    const int d = blockIdx.x;
    const unsigned short* Wih = d ? wih1 : wih0;
    const unsigned short* Whh = d ? whh1 : whh0;
    const float* bih = d ? bih1 : bih0;
    const float* bhh = d ? bhh1 : bhh0;

    const int tid = threadIdx.x, lane = tid & 31, wave = tid >> 5;  // 16 waves

    for (int i = tid; i < B_ * HE; i += 512) { h_f[i] = 0.f; h_b[i] = 0; }
    __syncthreads();

    for (int t = 0; t < S_; ++t) {
        int s = d ? (S_ - 1 - t) : t;
        const unsigned short* Xs = xs + (long)s * B_ * E;

        // GEMM phase: 96 tiles (2 mt x 48 nt), 6 tiles per wave
        for (int tau = wave; tau < 96; tau += 16) {
            int mt = tau / 48, nt = tau % 48;
            int m0 = mt * 16, n0 = nt * 16;
            v8f ai = {}, ah = {};
#pragma unroll
            for (int kb = 0; kb < E; kb += 32)
                ai = wmma_bf16(ai, load_a(Xs, E, m0, kb, lane),
                                   load_b(Wih, E, n0, kb, lane));
#pragma unroll
            for (int kb = 0; kb < HE; kb += 32)
                ah = wmma_bf16(ah, load_a(h_b, HE, m0, kb, lane),
                                   load_b(Whh, HE, n0, kb, lane));

            int n  = n0 + (lane & 15);
            int rb = m0 + ((lane >> 4) << 3);
            if (n0 < 2 * HE) {                       // r,z region: sum gi+gh
                float bs = bih[n] + bhh[n];
#pragma unroll
                for (int v = 0; v < 8; ++v)
                    g_rz[(rb + v) * 512 + n] = ai[v] + ah[v] + bs;
            } else {                                 // n region: keep separate
                int c = n - 2 * HE;
                float b1 = bih[n], b2 = bhh[n];
#pragma unroll
                for (int v = 0; v < 8; ++v) {
                    g_in[(rb + v) * HE + c] = ai[v] + b1;
                    g_hn[(rb + v) * HE + c] = ah[v] + b2;
                }
            }
        }
        __syncthreads();

        // pointwise GRU update
        for (int e = tid; e < B_ * HE; e += 512) {
            int m = e >> 8, c = e & 255;
            float r  = sigmoidf_(g_rz[m * 512 + c]);
            float z  = sigmoidf_(g_rz[m * 512 + 256 + c]);
            float nn = tanhf(g_in[e] + r * g_hn[e]);
            float hn = (1.f - z) * nn + z * h_f[e];
            h_f[e] = hn;
            h_b[e] = f2bf(hn);
            long o = ((long)m * S_ + s) * (2 * HE) + d * HE + c;
            enhy[o]    = hn;
            enhy_bf[o] = f2bf(hn);
        }
        __syncthreads();
    }
}

// ---------------- h0 = tanh([hT_f|hT_b] @ e2d_W.T + b) ----------------
__device__ inline v16bf load_a_hT(const unsigned short* enh, int m0, int kb, int lane) {
    int m  = m0 + (lane & 15);
    int k2 = kb + ((lane >> 4) << 3);
    long base = (long)m * S_ * 512;
    long addr = (k2 < 256) ? base + (long)(S_ - 1) * 512 + k2   // hT_f at s=S-1, cols 0..255
                           : base + k2;                          // hT_b at s=0,   cols 256..511
    const ushort8* p = (const ushort8*)(enh + addr);
    return mk16(p[0], p[2]);
}

__global__ __launch_bounds__(512) void k_h0(
    const unsigned short* __restrict__ enh_bf, const unsigned short* __restrict__ W,
    const float* __restrict__ bias, float* __restrict__ h0f, unsigned short* __restrict__ h0b)
{
    int tid = threadIdx.x, lane = tid & 31, wave = tid >> 5;
    for (int tau = wave; tau < 64; tau += 16) {       // 2 mt x 32 nt
        int mt = tau / 32, nt = tau % 32;
        int m0 = mt * 16, n0 = nt * 16;
        v8f acc = {};
#pragma unroll
        for (int kb = 0; kb < HD; kb += 32)
            acc = wmma_bf16(acc, load_a_hT(enh_bf, m0, kb, lane),
                                 load_b(W, HD, n0, kb, lane));
        int n = n0 + (lane & 15);
        int rb = m0 + ((lane >> 4) << 3);
        float bn = bias[n];
#pragma unroll
        for (int v = 0; v < 8; ++v) {
            float x = tanhf(acc[v] + bn);
            h0f[(rb + v) * HD + n] = x;
            h0b[(rb + v) * HD + n] = f2bf(x);
        }
    }
}

// ---------------- decoder: persistent, 1 block x 1024 threads ----------------
// LDS: h_f32(64KB) | h_bf16(32KB) | attn(51.2KB) | xcat_bf16(48KB) = 194KB
__global__ __launch_bounds__(1024) void k_decoder(
    const unsigned short* __restrict__ xt,    // (T,B,E) bf16
    const float* __restrict__ enhy,           // (B,S,512) f32
    const unsigned short* __restrict__ Wih, const unsigned short* __restrict__ Whh,
    const float* __restrict__ bih, const float* __restrict__ bhh,
    const float* __restrict__ h0f, const unsigned short* __restrict__ h0b,
    float* __restrict__ grz, float* __restrict__ gin, float* __restrict__ ghn, // L2 staging
    unsigned short* __restrict__ dec_out)     // (B,T,HD) bf16
{
    extern __shared__ char smem[];
    float*          h_f  = (float*)(smem);
    unsigned short* h_b  = (unsigned short*)(smem + 65536);
    float*          attn = (float*)(smem + 98304);
    unsigned short* xcat = (unsigned short*)(smem + 149504);

    const int tid = threadIdx.x, lane = tid & 31, wave = tid >> 5;  // 32 waves

    for (int i = tid; i < B_ * HD; i += 1024) { h_f[i] = h0f[i]; h_b[i] = h0b[i]; }
    __syncthreads();

    for (int t = 0; t < T_; ++t) {
        // 1) attention scores e[b,s] = enhy[b,s,:] . h[b,:]
        for (int e = tid; e < B_ * S_; e += 1024) {
            int b = e / S_, s = e % S_;
            const float4* eh = (const float4*)(enhy + ((long)b * S_ + s) * HD);
            const float4* hh = (const float4*)(h_f + b * HD);
            float acc = 0.f;
            for (int k = 0; k < HD / 4; ++k) {
                float4 x = eh[k], y = hh[k];
                acc += x.x * y.x + x.y * y.y + x.z * y.z + x.w * y.w;
            }
            attn[e] = acc;
        }
        __syncthreads();

        // 2) softmax over s (wave w handles batch b = w)
        {
            int b = wave;
            float mx = -1e30f;
            for (int s = lane; s < S_; s += 32) mx = fmaxf(mx, attn[b * S_ + s]);
            for (int o = 16; o > 0; o >>= 1) mx = fmaxf(mx, __shfl_xor(mx, o, 32));
            float sum = 0.f;
            for (int s = lane; s < S_; s += 32) {
                float v = __expf(attn[b * S_ + s] - mx);
                attn[b * S_ + s] = v;
                sum += v;
            }
            for (int o = 16; o > 0; o >>= 1) sum += __shfl_xor(sum, o, 32);
            float inv = 1.f / sum;
            for (int s = lane; s < S_; s += 32) attn[b * S_ + s] *= inv;
        }
        __syncthreads();

        // 3) ctx = tanh(attn @ enhy) -> xcat[:,256:768]; embed x -> xcat[:,0:256]
        for (int e = tid; e < B_ * HD; e += 1024) {
            int b = e >> 9, c = e & 511;
            const float* eh = enhy + (long)b * S_ * HD + c;
            const float* at = attn + b * S_;
            float acc = 0.f;
            for (int s = 0; s < S_; ++s) acc += at[s] * eh[(long)s * HD];
            xcat[b * 768 + 256 + c] = f2bf(tanhf(acc));
        }
        for (int e = tid; e < B_ * E; e += 1024) {
            int b = e >> 8, c = e & 255;
            xcat[b * 768 + c] = xt[((long)t * B_ + b) * E + c];
        }
        __syncthreads();

        // 4) GRU GEMMs: gi = xcat@Wih.T (K=768), gh = h@Whh.T (K=512); N=1536
        for (int tau = wave; tau < 192; tau += 32) {   // 2 mt x 96 nt, 6 tiles/wave
            int mt = tau / 96, nt = tau % 96;
            int m0 = mt * 16, n0 = nt * 16;
            v8f ai = {}, ah = {};
#pragma unroll 4
            for (int kb = 0; kb < 768; kb += 32)
                ai = wmma_bf16(ai, load_a(xcat, 768, m0, kb, lane),
                                   load_b(Wih, 768, n0, kb, lane));
#pragma unroll 4
            for (int kb = 0; kb < HD; kb += 32)
                ah = wmma_bf16(ah, load_a(h_b, HD, m0, kb, lane),
                                   load_b(Whh, HD, n0, kb, lane));

            int n  = n0 + (lane & 15);
            int rb = m0 + ((lane >> 4) << 3);
            if (n0 < 2 * HD) {
                float bs = bih[n] + bhh[n];
#pragma unroll
                for (int v = 0; v < 8; ++v)
                    grz[(rb + v) * 1024 + n] = ai[v] + ah[v] + bs;
            } else {
                int c = n - 2 * HD;
                float b1 = bih[n], b2 = bhh[n];
#pragma unroll
                for (int v = 0; v < 8; ++v) {
                    gin[(rb + v) * HD + c] = ai[v] + b1;
                    ghn[(rb + v) * HD + c] = ah[v] + b2;
                }
            }
        }
        __threadfence();
        __syncthreads();

        // 5) pointwise GRU update + dec_out write
        for (int e = tid; e < B_ * HD; e += 1024) {
            int m = e >> 9, c = e & 511;
            float r  = sigmoidf_(grz[m * 1024 + c]);
            float z  = sigmoidf_(grz[m * 1024 + 512 + c]);
            float nn = tanhf(gin[e] + r * ghn[e]);
            float hn = (1.f - z) * nn + z * h_f[e];
            h_f[e] = hn;
            h_b[e] = f2bf(hn);
            dec_out[((long)m * T_ + t) * HD + c] = f2bf(hn);
        }
        __syncthreads();
    }
}

// ---------------- logits GEMM: (B*T,512) x (512,32000), 2x2 register blocking ----
// Block: 256 threads = 8 waves. Wave computes a 32x32 output block (2 m-tiles x
// 2 n-tiles) with 4 accumulators; each K-step loads 2 A + 2 B fragments feeding
// 4 WMMAs (each fragment reused twice).
__global__ __launch_bounds__(256) void k_logits(
    const unsigned short* __restrict__ A,     // (B*T, HD) bf16
    const unsigned short* __restrict__ W,     // (V, HD)   bf16
    const float* __restrict__ bias, float* __restrict__ out)
{
    const int lane = threadIdx.x & 31, wave = threadIdx.x >> 5;
    const int m0    = blockIdx.y * 32;                    // 2 m-tiles
    const int nbase = (blockIdx.x * 16 + wave * 2) * 16;  // 2 n-tiles
    v8f acc00 = {}, acc01 = {}, acc10 = {}, acc11 = {};
#pragma unroll 4
    for (int kb = 0; kb < HD; kb += 32) {
        if (kb < HD - 64) {
            __builtin_prefetch(W + (long)(nbase + (lane & 15)) * HD + kb + 64, 0, 1);
            __builtin_prefetch(W + (long)(nbase + 16 + (lane & 15)) * HD + kb + 64, 0, 1);
        }
        v16bf a0 = load_a(A, HD, m0,      kb, lane);
        v16bf a1 = load_a(A, HD, m0 + 16, kb, lane);
        v16bf b0 = load_b(W, HD, nbase,      kb, lane);
        v16bf b1 = load_b(W, HD, nbase + 16, kb, lane);
        acc00 = wmma_bf16(acc00, a0, b0);
        acc01 = wmma_bf16(acc01, a0, b1);
        acc10 = wmma_bf16(acc10, a1, b0);
        acc11 = wmma_bf16(acc11, a1, b1);
    }
    const int nlo = lane & 15;
    const int rlo = (lane >> 4) << 3;
    float bn0 = bias[nbase + nlo];
    float bn1 = bias[nbase + 16 + nlo];
#pragma unroll
    for (int v = 0; v < 8; ++v) {
        long r0 = (long)(m0 + rlo + v) * VOCAB;
        long r1 = (long)(m0 + 16 + rlo + v) * VOCAB;
        out[r0 + nbase + nlo]      = acc00[v] + bn0;
        out[r0 + nbase + 16 + nlo] = acc01[v] + bn1;
        out[r1 + nbase + nlo]      = acc10[v] + bn0;
        out[r1 + nbase + 16 + nlo] = acc11[v] + bn1;
    }
}

// ---------------- launch ----------------
extern "C" void kernel_launch(void* const* d_in, const int* in_sizes, int n_in,
                              void* d_out, int out_size, void* d_ws, size_t ws_size,
                              hipStream_t stream) {
    (void)in_sizes; (void)n_in; (void)out_size; (void)ws_size;

    const int*   src_ids   = (const int*)d_in[0];
    const int*   trg_ids   = (const int*)d_in[1];
    const float* src_emb   = (const float*)d_in[2];
    const float* trg_emb   = (const float*)d_in[3];
    const float* eWih_f    = (const float*)d_in[4];
    const float* eWhh_f    = (const float*)d_in[5];
    const float* ebih_f    = (const float*)d_in[6];
    const float* ebhh_f    = (const float*)d_in[7];
    const float* eWih_b    = (const float*)d_in[8];
    const float* eWhh_b    = (const float*)d_in[9];
    const float* ebih_b    = (const float*)d_in[10];
    const float* ebhh_b    = (const float*)d_in[11];
    const float* e2d_W     = (const float*)d_in[12];
    const float* e2d_b     = (const float*)d_in[13];
    const float* dWih      = (const float*)d_in[14];
    const float* dWhh      = (const float*)d_in[15];
    const float* dbih      = (const float*)d_in[16];
    const float* dbhh      = (const float*)d_in[17];
    const float* out_W     = (const float*)d_in[18];
    const float* out_b     = (const float*)d_in[19];
    float* logits = (float*)d_out;

    char* ws = (char*)d_ws;
    size_t off = 0;
    auto alloc = [&](size_t bytes) {
        void* p = ws + off;
        off += (bytes + 255) & ~(size_t)255;
        return p;
    };
    unsigned short* xs_bf   = (unsigned short*)alloc((size_t)S_ * B_ * E * 2);
    unsigned short* xt_bf   = (unsigned short*)alloc((size_t)T_ * B_ * E * 2);
    unsigned short* wih_f   = (unsigned short*)alloc((size_t)3 * HE * E * 2);
    unsigned short* whh_f   = (unsigned short*)alloc((size_t)3 * HE * HE * 2);
    unsigned short* wih_b   = (unsigned short*)alloc((size_t)3 * HE * E * 2);
    unsigned short* whh_b   = (unsigned short*)alloc((size_t)3 * HE * HE * 2);
    unsigned short* we2d    = (unsigned short*)alloc((size_t)HD * 2 * HE * 2);
    unsigned short* wdih    = (unsigned short*)alloc((size_t)3 * HD * (E + HD) * 2);
    unsigned short* wdhh    = (unsigned short*)alloc((size_t)3 * HD * HD * 2);
    unsigned short* wout    = (unsigned short*)alloc((size_t)VOCAB * HD * 2);
    float*          enhy    = (float*)alloc((size_t)B_ * S_ * 2 * HE * 4);
    unsigned short* enhy_bf = (unsigned short*)alloc((size_t)B_ * S_ * 2 * HE * 2);
    float*          hdec_f  = (float*)alloc((size_t)B_ * HD * 4);
    unsigned short* hdec_b  = (unsigned short*)alloc((size_t)B_ * HD * 2);
    float*          grz     = (float*)alloc((size_t)B_ * 2 * HD * 4);
    float*          gin     = (float*)alloc((size_t)B_ * HD * 4);
    float*          ghn     = (float*)alloc((size_t)B_ * HD * 4);
    unsigned short* dec_out = (unsigned short*)alloc((size_t)B_ * T_ * HD * 2);

    auto cvt = [&](const float* s, unsigned short* d, long n) {
        int blocks = (int)((n + 255) / 256);
        if (blocks > 65535) blocks = 65535;
        k_f32_to_bf16<<<blocks, 256, 0, stream>>>(s, d, n);
    };
    cvt(eWih_f, wih_f, (long)3 * HE * E);
    cvt(eWhh_f, whh_f, (long)3 * HE * HE);
    cvt(eWih_b, wih_b, (long)3 * HE * E);
    cvt(eWhh_b, whh_b, (long)3 * HE * HE);
    cvt(e2d_W,  we2d,  (long)HD * 2 * HE);
    cvt(dWih,   wdih,  (long)3 * HD * (E + HD));
    cvt(dWhh,   wdhh,  (long)3 * HD * HD);
    cvt(out_W,  wout,  (long)VOCAB * HD);

    {
        long n = (long)S_ * B_ * E;
        k_embed<<<(int)((n + 255) / 256), 256, 0, stream>>>(src_ids, src_emb, xs_bf, S_, n);
    }
    {
        long n = (long)T_ * B_ * E;
        k_embed<<<(int)((n + 255) / 256), 256, 0, stream>>>(trg_ids, trg_emb, xt_bf, T_, n);
    }

    k_encoder<<<2, 512, 180224, stream>>>(xs_bf,
        wih_f, whh_f, ebih_f, ebhh_f,
        wih_b, whh_b, ebih_b, ebhh_b,
        enhy, enhy_bf);

    k_h0<<<1, 512, 0, stream>>>(enhy_bf, we2d, e2d_b, hdec_f, hdec_b);

    k_decoder<<<1, 1024, 198656, stream>>>(xt_bf, enhy,
        wdih, wdhh, dbih, dbhh,
        hdec_f, hdec_b, grz, gin, ghn, dec_out);

    // 32000 cols / (8 waves * 2 n-tiles * 16) = 125; 3200 rows / 32 = 100
    dim3 g(VOCAB / 256, (B_ * T_) / 32);
    k_logits<<<g, 256, 0, stream>>>(dec_out, wout, out_b, logits);
}
